// RecommendationModel_16432544874513
// MI455X (gfx1250) — matrix operand order repaired and checked by reading, not compiled, not run on hardware
//
#include <hip/hip_runtime.h>
#include <math.h>

// ---------------------------------------------------------------------------
// Model constants (from reference)
// ---------------------------------------------------------------------------
#define HIDN 128
#define NHEAD 2
#define DHEAD 64
#define NUC 20000
#define NIC 8192
#define NINGC 100000
#define EPOC 300000
#define EBC 400000
#define LCLC 2
#define LSC 3
#define LFC 3

typedef __bf16 bf16;
typedef __attribute__((ext_vector_type(16))) __bf16 v16bf;
typedef __attribute__((ext_vector_type(8)))  float  v8f;

__device__ __forceinline__ float gelu_f(float x) {
    return 0.5f * x * (1.0f + erff(x * 0.70710678118654752f));
}
// order-preserving float <-> uint key for atomicMax-based segment max
__device__ __forceinline__ unsigned fkey(float f) {
    unsigned u = __float_as_uint(f);
    return (u & 0x80000000u) ? ~u : (u | 0x80000000u);
}
__device__ __forceinline__ float fkeyinv(unsigned k) {
    return (k & 0x80000000u) ? __uint_as_float(k & 0x7FFFFFFFu) : __uint_as_float(~k);
}

// ---------------------------------------------------------------------------
// Generic bf16 WMMA GEMM:  C[M x Ncols] = epi( actA(A[M x K]) @ W[K x Ncols] + bias )
// block = 256 threads = 8 waves; wave w -> rows [bx*128 + w*16, +16), cols [by*16,+16)
// epi: 0 = none, 1 = relu, 2 = skip-mix  s*val + (1-s)*Xres
// ---------------------------------------------------------------------------
__global__ void gemm_wmma_kernel(const float* __restrict__ A, int lda, int geluA,
                                 const float* __restrict__ W, int ldw,
                                 const float* __restrict__ bias,
                                 float* __restrict__ C, int ldc,
                                 int M, int K, int epi,
                                 const float* __restrict__ skip,
                                 const float* __restrict__ Xres, int ldx) {
    int wave = threadIdx.x >> 5;
    int lane = threadIdx.x & 31;
    int rowBase = blockIdx.x * 128 + wave * 16;
    if (rowBase >= M) return;
    int colBase = blockIdx.y * 16;
    int mrow = rowBase + (lane & 15);
    int hiHalf = lane >> 4;               // 0 or 1
    v8f acc = {};
    for (int k0 = 0; k0 < K; k0 += 32) {
        v16bf a, b;
#pragma unroll
        for (int h = 0; h < 16; ++h) {     // A: 16x32, lane=M%16, K per ISA layout
            int r = h >> 1;
            int k = ((r & 4) << 2) + (hiHalf << 3) + ((r & 3) << 1) + (h & 1);
            float av = A[(size_t)mrow * lda + k0 + k];
            if (geluA) av = gelu_f(av);
            a[h] = (bf16)av;
        }
        int n = colBase + (lane & 15);
#pragma unroll
        for (int h = 0; h < 16; ++h) {     // B: 32x16, lane=N, K per ISA layout (same as A)
            int k = ((h & 8) << 1) + (hiHalf << 3) + (h & 7);
            b[h] = (bf16)W[(size_t)(k0 + k) * ldw + n];
        }
        acc = __builtin_amdgcn_wmma_f32_16x16x32_bf16(false, a, false, b,
                                                      (short)0, acc, false, false);
    }
    int col = colBase + (lane & 15);
    float bv = bias ? bias[col] : 0.0f;
    float sgate = 0.0f;
    if (epi == 2) sgate = 1.0f / (1.0f + __expf(-skip[0]));
#pragma unroll
    for (int r = 0; r < 8; ++r) {          // C: VGPR r -> row r (+8 for hi half)
        int row = rowBase + r + (hiHalf << 3);
        float v = acc[r] + bv;
        if (epi == 1) v = fmaxf(v, 0.0f);
        else if (epi == 2) v = sgate * v + (1.0f - sgate) * Xres[(size_t)row * ldx + col];
        C[(size_t)row * ldc + col] = v;
    }
}

// ---------------------------------------------------------------------------
// Fused WMMA logsumexp over logits = TN @ UN^T (TN pre-scaled by 1/TEMP).
// One block per 16 rows; 8 waves stream 16-col tiles with online max/sum.
// ---------------------------------------------------------------------------
__global__ void lse_wmma_kernel(const float* __restrict__ TN,
                                const float* __restrict__ UN,
                                float* __restrict__ lse, int N) {
    __shared__ float lm[8][16], ls[8][16];
    int wave = threadIdx.x >> 5, lane = threadIdx.x & 31;
    int ibase = blockIdx.x * 16;
    int mrow = ibase + (lane & 15);
    int hiHalf = lane >> 4;
    v16bf afrag[4];
    for (int ks = 0; ks < 4; ++ks) {
#pragma unroll
        for (int h = 0; h < 16; ++h) {
            int r = h >> 1;
            int k = ks * 32 + ((r & 4) << 2) + (hiHalf << 3) + ((r & 3) << 1) + (h & 1);
            afrag[ks][h] = (bf16)TN[(size_t)mrow * HIDN + k];
        }
    }
    float runm[8], runs[8];
#pragma unroll
    for (int r = 0; r < 8; ++r) { runm[r] = -3.0e38f; runs[r] = 0.0f; }
    int nblocks = N >> 4;
    for (int jb = wave; jb < nblocks; jb += 8) {
        int j = jb * 16 + (lane & 15);
        v8f acc = {};
        for (int ks = 0; ks < 4; ++ks) {
            v16bf b;
#pragma unroll
            for (int h = 0; h < 16; ++h) {   // B[k][n] = UN[n][k]
                int k = ks * 32 + ((h & 8) << 1) + (hiHalf << 3) + (h & 7);
                b[h] = (bf16)UN[(size_t)j * HIDN + k];
            }
            acc = __builtin_amdgcn_wmma_f32_16x16x32_bf16(false, afrag[ks], false, b,
                                                          (short)0, acc, false, false);
        }
#pragma unroll
        for (int r = 0; r < 8; ++r) {
            float v = acc[r];
            if (v > runm[r]) { runs[r] = runs[r] * __expf(runm[r] - v) + 1.0f; runm[r] = v; }
            else             runs[r] += __expf(v - runm[r]);
        }
    }
#pragma unroll
    for (int r = 0; r < 8; ++r) {
        for (int off = 8; off >= 1; off >>= 1) {   // stays inside 16-lane half
            float om = __shfl_xor(runm[r], off, 32);
            float os = __shfl_xor(runs[r], off, 32);
            float nm = fmaxf(runm[r], om);
            runs[r] = runs[r] * __expf(runm[r] - nm) + os * __expf(om - nm);
            runm[r] = nm;
        }
    }
    if ((lane & 15) == 0) {
#pragma unroll
        for (int r = 0; r < 8; ++r) {
            int row = r + (hiHalf << 3);
            lm[wave][row] = runm[r];
            ls[wave][row] = runs[r];
        }
    }
    __syncthreads();
    if (threadIdx.x < 16) {
        float m = -3.0e38f;
        for (int w = 0; w < 8; ++w) m = fmaxf(m, lm[w][threadIdx.x]);
        float s = 0.0f;
        for (int w = 0; w < 8; ++w) s += ls[w][threadIdx.x] * __expf(lm[w][threadIdx.x] - m);
        lse[ibase + threadIdx.x] = m + __logf(s);
    }
}

// ---------------------------------------------------------------------------
// Small utility kernels
// ---------------------------------------------------------------------------
__global__ void slice_copy_kernel(const float* __restrict__ in, int ldin,
                                  float* __restrict__ out, int N) {
    size_t idx = (size_t)blockIdx.x * blockDim.x + threadIdx.x;
    if (idx >= (size_t)N * HIDN) return;
    size_t i = idx >> 7; int c = (int)(idx & 127);
    out[idx] = in[i * (size_t)ldin + c];
}
__global__ void gather_rows_kernel(const float* __restrict__ emb, const int* __restrict__ ids,
                                   float* __restrict__ out, int N) {
    size_t idx = (size_t)blockIdx.x * blockDim.x + threadIdx.x;
    if (idx >= (size_t)N * HIDN) return;
    size_t i = idx >> 7; int c = (int)(idx & 127);
    out[idx] = emb[(size_t)ids[i] * HIDN + c];
}
__global__ void rownorm_kernel(const float* __restrict__ X, float* __restrict__ Y,
                               int N, float scale) {
    int row = blockIdx.x * 8 + (threadIdx.x >> 5);
    int lane = threadIdx.x & 31;
    if (row >= N) return;
    const float* x = X + (size_t)row * HIDN;
    float ss = 0.f;
    for (int c = lane; c < HIDN; c += 32) { float v = x[c]; ss += v * v; }
    for (int off = 16; off; off >>= 1) ss += __shfl_xor(ss, off, 32);
    float inv = scale / fmaxf(sqrtf(ss), 1e-12f);
    float* y = Y + (size_t)row * HIDN;
    for (int c = lane; c < HIDN; c += 32) y[c] = x[c] * inv;
}
__global__ void rowdot_kernel(const float* __restrict__ X, const float* __restrict__ v,
                              float* __restrict__ out, int N) {
    int row = blockIdx.x * 8 + (threadIdx.x >> 5);
    int lane = threadIdx.x & 31;
    if (row >= N) return;
    float s = 0.f;
    for (int c = lane; c < HIDN; c += 32) s += X[(size_t)row * HIDN + c] * v[c];
    for (int off = 16; off; off >>= 1) s += __shfl_xor(s, off, 32);
    if (lane == 0) out[row] = s;
}
__global__ void pairdot_kernel(const float* __restrict__ Xa, const float* __restrict__ Xb,
                               float* __restrict__ out, int N) {
    int row = blockIdx.x * 8 + (threadIdx.x >> 5);
    int lane = threadIdx.x & 31;
    if (row >= N) return;
    float s = 0.f;
    for (int c = lane; c < HIDN; c += 32)
        s += Xa[(size_t)row * HIDN + c] * Xb[(size_t)row * HIDN + c];
    for (int off = 16; off; off >>= 1) s += __shfl_xor(s, off, 32);
    if (lane == 0) out[row] = s;
}
__global__ void nutproj_kernel(const float* __restrict__ nut, const float* __restrict__ W,
                               const float* __restrict__ b, float* __restrict__ out) {
    size_t idx = (size_t)blockIdx.x * blockDim.x + threadIdx.x;
    if (idx >= (size_t)NIC * HIDN) return;
    size_t i = idx >> 7; int c = (int)(idx & 127);
    float s = b[c];
    for (int k = 0; k < 20; ++k) s += nut[i * 20 + k] * W[k * HIDN + c];
    out[idx] = s;
}
__global__ void lossred_kernel(const float* __restrict__ lse, const float* __restrict__ diag,
                               float* __restrict__ cl, float scale, int N) {
    __shared__ float sh[256];
    float s = 0.f;
    for (int i = blockIdx.x * 256 + threadIdx.x; i < N; i += gridDim.x * 256)
        s += lse[i] - diag[i];
    sh[threadIdx.x] = s; __syncthreads();
    for (int o = 128; o > 0; o >>= 1) {
        if (threadIdx.x < o) sh[threadIdx.x] += sh[threadIdx.x + o];
        __syncthreads();
    }
    if (threadIdx.x == 0) atomicAdd(cl, sh[0] * scale);
}
// column-wise batchnorm statistics (biased var); one block per column
__global__ void bnstats_kernel(const float* __restrict__ X, const float* __restrict__ R,
                               int reluX, int N, float* __restrict__ mv) {
    __shared__ float shs[256], shq[256];
    int c = blockIdx.x;
    float s = 0.f, q = 0.f;
    for (int i = threadIdx.x; i < N; i += 256) {
        float v = X[(size_t)i * HIDN + c];
        if (reluX) v = fmaxf(v, 0.f);
        if (R) v += R[(size_t)i * HIDN + c];
        s += v; q += v * v;
    }
    shs[threadIdx.x] = s; shq[threadIdx.x] = q;
    __syncthreads();
    for (int o = 128; o > 0; o >>= 1) {
        if (threadIdx.x < o) { shs[threadIdx.x] += shs[threadIdx.x + o];
                               shq[threadIdx.x] += shq[threadIdx.x + o]; }
        __syncthreads();
    }
    if (threadIdx.x == 0) {
        float m = shs[0] / (float)N;
        mv[c] = m;
        mv[HIDN + c] = shq[0] / (float)N - m * m;
    }
}
__global__ void bnapply_kernel(const float* __restrict__ X, const float* __restrict__ R,
                               int reluX, const float* __restrict__ mv,
                               const float* __restrict__ g, const float* __restrict__ b,
                               float* __restrict__ Y, int relu, size_t total) {
    size_t idx = (size_t)blockIdx.x * blockDim.x + threadIdx.x;
    if (idx >= total) return;
    int c = (int)(idx & 127);
    float v = X[idx];
    if (reluX) v = fmaxf(v, 0.f);
    if (R) v += R[idx];
    v = (v - mv[c]) * rsqrtf(mv[HIDN + c] + 1e-5f) * g[c] + b[c];
    if (relu) v = fmaxf(v, 0.f);
    Y[idx] = v;
}
// ---------------------------------------------------------------------------
// Edge kernels (segment softmax via atomic max-key / sum, then scatter)
// ---------------------------------------------------------------------------
__global__ void han_edge_kernel(const int* __restrict__ src, const int* __restrict__ dst,
                                const float* __restrict__ si, const float* __restrict__ dj,
                                float* __restrict__ alpha, unsigned* __restrict__ mx, int E) {
    int e = blockIdx.x * blockDim.x + threadIdx.x;
    if (e >= E) return;
    float a = si[src[e]] + dj[dst[e]];
    a = (a > 0.f) ? a : 0.2f * a;          // leaky_relu 0.2
    alpha[e] = a;
    atomicMax(&mx[dst[e]], fkey(a));
}
__global__ void seg_expsum_kernel(const float* __restrict__ alpha, const int* __restrict__ dst,
                                  const unsigned* __restrict__ mx, float* __restrict__ ex,
                                  float* __restrict__ sm, int E, int Hh) {
    int i = blockIdx.x * blockDim.x + threadIdx.x;
    if (i >= E * Hh) return;
    int e = i / Hh, h = i - e * Hh;
    int d = dst[e] * Hh + h;
    float v = __expf(alpha[i] - fkeyinv(mx[d]));
    ex[i] = v;
    atomicAdd(&sm[d], v);
}
__global__ void han_scatter_kernel(const int* __restrict__ src, const int* __restrict__ dst,
                                   const float* __restrict__ hi, const float* __restrict__ ex,
                                   const float* __restrict__ sm, float* __restrict__ out, int E) {
    size_t idx = (size_t)blockIdx.x * blockDim.x + threadIdx.x;
    if (idx >= (size_t)E * HIDN) return;
    int e = (int)(idx >> 7), c = (int)(idx & 127);
    int d = dst[e];
    float w = ex[e] / (sm[d] + 1e-16f);
    atomicAdd(&out[(size_t)d * HIDN + c], hi[(size_t)src[e] * HIDN + c] * w);
}
__global__ void hgt_alpha_kernel(const int* __restrict__ src, const int* __restrict__ dst,
                                 const float* __restrict__ Q, const float* __restrict__ KR,
                                 const float* __restrict__ prel, float* __restrict__ alpha,
                                 unsigned* __restrict__ mx, int E, float scale) {
    int i = blockIdx.x * blockDim.x + threadIdx.x;
    if (i >= E * 2) return;
    int e = i >> 1, h = i & 1;
    const float* q = Q + (size_t)dst[e] * HIDN + h * DHEAD;
    const float* k = KR + (size_t)src[e] * HIDN + h * DHEAD;
    float acc = 0.f;
    for (int t = 0; t < DHEAD; ++t) acc += q[t] * k[t];
    float a = acc * prel[h] * scale;
    alpha[i] = a;
    atomicMax(&mx[dst[e] * 2 + h], fkey(a));
}
__global__ void hgt_scatter_kernel(const int* __restrict__ src, const int* __restrict__ dst,
                                   const float* __restrict__ VRp, const float* __restrict__ ex,
                                   const float* __restrict__ sm, float* __restrict__ agg, int E) {
    size_t idx = (size_t)blockIdx.x * blockDim.x + threadIdx.x;
    if (idx >= (size_t)E * HIDN) return;
    int e = (int)(idx >> 7), c = (int)(idx & 127), h = c >> 6;
    int d = dst[e];
    float w = ex[e * 2 + h] / (sm[d * 2 + h] + 1e-16f);
    atomicAdd(&agg[(size_t)d * HIDN + c], VRp[(size_t)src[e] * HIDN + c] * w);
}

// ---------------------------------------------------------------------------
// Host orchestration
// ---------------------------------------------------------------------------
extern "C" void kernel_launch(void* const* d_in, const int* in_sizes, int n_in,
                              void* d_out, int out_size, void* d_ws, size_t ws_size,
                              hipStream_t stream) {
    (void)in_sizes; (void)n_in; (void)out_size; (void)ws_size;
    auto Ip = [&](int i) { return (const int*)d_in[i]; };
    auto Fp = [&](int i) { return (const float*)d_in[i]; };

    // output slices (evolving node states live here)
    float* OUTp   = (float*)d_out;
    float* USER   = OUTp;
    float* ITEM   = USER   + (size_t)NUC * HIDN;
    float* TASTE  = ITEM   + (size_t)NIC * HIDN;
    float* INTENT = TASTE  + (size_t)NIC * HIDN;
    float* IMAGE  = INTENT + (size_t)NIC * HIDN;
    float* INGRED = IMAGE  + (size_t)NIC * HIDN;
    float* CL     = INGRED + (size_t)NINGC * HIDN;

    // workspace layout
    const size_t R_TOT = (size_t)NUC + 4 * (size_t)NIC;  // 52768 rows (all types)
    float* wsp = (float*)d_ws;
    size_t wo = 0;
    auto alloc = [&](size_t n) { float* p = wsp + wo; wo += n; return p; };
    float* HI   = alloc((size_t)NINGC * HIDN);
    float* HT   = alloc((size_t)NIC * HIDN);
    float* TMPI = alloc((size_t)NIC * HIDN);
    float* TN   = alloc((size_t)NIC * HIDN);
    float* UNb  = alloc((size_t)NIC * HIDN);
    float* Kb   = alloc(R_TOT * HIDN);
    float* Qb   = alloc(R_TOT * HIDN);
    float* Vb   = alloc(R_TOT * HIDN);
    float* KRb  = alloc(R_TOT * HIDN);
    float* VRb  = alloc(R_TOT * HIDN);
    float* AGGU = alloc((size_t)NUC * HIDN);
    float* AGGI = alloc((size_t)NIC * HIDN);
    float* OU   = alloc((size_t)NUC * HIDN);
    float* SIb  = alloc(NINGC);
    float* DJb  = alloc(NIC);
    float* LSEb = alloc(NIC);
    float* DIAG = alloc(NIC);
    float* EA   = alloc(1700000);
    float* EX   = alloc(1700000);
    unsigned* MXI = (unsigned*)alloc(NIC * 2);
    float*    SMI = alloc(NIC * 2);
    unsigned* MXU = (unsigned*)alloc(NUC * 2);
    float*    SMU = alloc(NUC * 2);
    float*    MV  = alloc(2 * HIDN);

    auto g1 = [&](size_t n) { return dim3((unsigned)((n + 255) / 256)); };
    auto gemm = [&](const float* A, int lda, int geluA, const float* W, int ldw,
                    const float* bias, float* C, int ldc, int M, int Ncols, int K,
                    int epi, const float* skip, const float* Xres, int ldx) {
        dim3 grid((unsigned)((M + 127) / 128), (unsigned)(Ncols / 16));
        gemm_wmma_kernel<<<grid, 256, 0, stream>>>(A, lda, geluA, W, ldw, bias,
                                                   C, ldc, M, K, epi, skip, Xres, ldx);
    };

    // ---- 0. initialize node states ----------------------------------------
    gather_rows_kernel<<<g1((size_t)NUC * HIDN), 256, 0, stream>>>(Fp(19), Ip(0), USER, NUC);
    slice_copy_kernel<<<g1((size_t)NIC * HIDN), 256, 0, stream>>>(Fp(18), 128,  ITEM,   NIC);
    slice_copy_kernel<<<g1((size_t)NIC * HIDN), 256, 0, stream>>>(Fp(17), 384,  TASTE,  NIC);
    slice_copy_kernel<<<g1((size_t)NIC * HIDN), 256, 0, stream>>>(Fp(14), 384,  INTENT, NIC);
    slice_copy_kernel<<<g1((size_t)NIC * HIDN), 256, 0, stream>>>(Fp(13), 1024, IMAGE,  NIC);
    slice_copy_kernel<<<g1((size_t)NINGC * HIDN), 256, 0, stream>>>(Fp(16), 384, INGRED, NINGC);

    // ---- 1. InfoNCE contrastive loss (fused WMMA logsumexp) ---------------
    hipMemsetAsync(CL, 0, sizeof(float), stream);
    rownorm_kernel<<<dim3(NIC / 8), 256, 0, stream>>>(INTENT, TN, NIC, 1.0f / 0.05f);
    for (int l = 0; l < LCLC; ++l) {
        nutproj_kernel<<<g1((size_t)NIC * HIDN), 256, 0, stream>>>(
            Fp(15), Fp(20) + (size_t)l * 20 * HIDN, Fp(21) + (size_t)l * HIDN, TMPI);
        rownorm_kernel<<<dim3(NIC / 8), 256, 0, stream>>>(TMPI, UNb, NIC, 1.0f);
        pairdot_kernel<<<dim3(NIC / 8), 256, 0, stream>>>(TN, UNb, DIAG, NIC);
        lse_wmma_kernel<<<dim3(NIC / 16), 256, 0, stream>>>(TN, UNb, LSEb, NIC);
        lossred_kernel<<<dim3(64), 256, 0, stream>>>(LSEb, DIAG, CL,
                                                     1.0f / ((float)NIC * LCLC), NIC);
    }
    // intention = bn(intention, cl_g, cl_b)   (no relu)
    bnstats_kernel<<<dim3(HIDN), 256, 0, stream>>>(INTENT, nullptr, 0, NIC, MV);
    bnapply_kernel<<<g1((size_t)NIC * HIDN), 256, 0, stream>>>(
        INTENT, nullptr, 0, MV, Fp(22), Fp(23), INTENT, 0, (size_t)NIC * HIDN);

    // ---- 2. HAN layers ----------------------------------------------------
    for (int l = 0; l < LSC; ++l) {
        gemm(INGRED, HIDN, 0, Fp(24) + (size_t)l * HIDN * HIDN, HIDN,
             Fp(25) + (size_t)l * HIDN, HI, HIDN, NINGC, HIDN, HIDN, 0, nullptr, nullptr, 0);
        gemm(TASTE, HIDN, 0, Fp(26) + (size_t)l * HIDN * HIDN, HIDN,
             Fp(27) + (size_t)l * HIDN, HT, HIDN, NIC, HIDN, HIDN, 0, nullptr, nullptr, 0);
        rowdot_kernel<<<dim3((NINGC + 7) / 8), 256, 0, stream>>>(HI, Fp(28) + (size_t)l * HIDN, SIb, NINGC);
        rowdot_kernel<<<dim3(NIC / 8), 256, 0, stream>>>(HT, Fp(29) + (size_t)l * HIDN, DJb, NIC);
        hipMemsetAsync(MXI, 0, NIC * sizeof(unsigned), stream);
        hipMemsetAsync(SMI, 0, NIC * sizeof(float), stream);
        han_edge_kernel<<<g1(EPOC), 256, 0, stream>>>(Ip(1), Ip(2), SIb, DJb, EA, MXI, EPOC);
        seg_expsum_kernel<<<g1(EPOC), 256, 0, stream>>>(EA, Ip(2), MXI, EX, SMI, EPOC, 1);
        hipMemsetAsync(TMPI, 0, (size_t)NIC * HIDN * sizeof(float), stream);
        han_scatter_kernel<<<g1((size_t)EPOC * HIDN), 256, 0, stream>>>(Ip(1), Ip(2), HI, EX, SMI, TMPI, EPOC);
        // taste = relu(bn(relu(t) + taste))
        bnstats_kernel<<<dim3(HIDN), 256, 0, stream>>>(TMPI, TASTE, 1, NIC, MV);
        bnapply_kernel<<<g1((size_t)NIC * HIDN), 256, 0, stream>>>(
            TMPI, TASTE, 1, MV, Fp(30) + (size_t)l * HIDN, Fp(31) + (size_t)l * HIDN,
            TASTE, 1, (size_t)NIC * HIDN);
    }
    // sensory BNs
    bnstats_kernel<<<dim3(HIDN), 256, 0, stream>>>(TASTE, nullptr, 0, NIC, MV);
    bnapply_kernel<<<g1((size_t)NIC * HIDN), 256, 0, stream>>>(
        TASTE, nullptr, 0, MV, Fp(32), Fp(33), TASTE, 1, (size_t)NIC * HIDN);
    bnstats_kernel<<<dim3(HIDN), 256, 0, stream>>>(INGRED, nullptr, 0, NINGC, MV);
    bnapply_kernel<<<g1((size_t)NINGC * HIDN), 256, 0, stream>>>(
        INGRED, nullptr, 0, MV, Fp(32) + HIDN, Fp(33) + HIDN, INGRED, 1, (size_t)NINGC * HIDN);

    // ---- 3. HGT layers ----------------------------------------------------
    const size_t toff[5] = { 0, (size_t)NUC, (size_t)NUC + NIC,
                             (size_t)NUC + 2 * NIC, (size_t)NUC + 3 * NIC }; // user,item,taste,intention,image
    float* Xt[5] = { USER, ITEM, TASTE, INTENT, IMAGE };
    int    Mt[5] = { NUC, NIC, NIC, NIC, NIC };
    const int relS[5]  = { 2, 3, 4, 0, 1 };     // stype index per relation (taste,intention,image,user | item)
    const int relSi[5] = { 3, 5, 7, 9, 11 };    // src input index
    const int relDi[5] = { 4, 6, 8, 10, 12 };   // dst input index
    const int relE[5]  = { NIC, NIC, NIC, EBC, EBC };

    for (int l = 0; l < LFC; ++l) {
        // K/Q/V projections per node type
        for (int t = 0; t < 5; ++t) {
            size_t wi = ((size_t)l * 5 + t);
            gemm(Xt[t], HIDN, 0, Fp(34) + wi * HIDN * HIDN, HIDN, Fp(35) + wi * HIDN,
                 Kb + toff[t] * HIDN, HIDN, Mt[t], HIDN, HIDN, 0, nullptr, nullptr, 0);
            gemm(Xt[t], HIDN, 0, Fp(36) + wi * HIDN * HIDN, HIDN, Fp(37) + wi * HIDN,
                 Qb + toff[t] * HIDN, HIDN, Mt[t], HIDN, HIDN, 0, nullptr, nullptr, 0);
            gemm(Xt[t], HIDN, 0, Fp(38) + wi * HIDN * HIDN, HIDN, Fp(39) + wi * HIDN,
                 Vb + toff[t] * HIDN, HIDN, Mt[t], HIDN, HIDN, 0, nullptr, nullptr, 0);
        }
        // per-relation per-head 64x64 relation transforms (k_rel, v_rel)
        for (int e = 0; e < 5; ++e) {
            int st = relS[e];
            size_t off = toff[st] * HIDN;
            for (int h = 0; h < NHEAD; ++h) {
                size_t wi = (((size_t)l * 5 + e) * NHEAD + h) * DHEAD * DHEAD;
                gemm(Kb + off + h * DHEAD, HIDN, 0, Fp(43) + wi, DHEAD, nullptr,
                     KRb + off + h * DHEAD, HIDN, Mt[st], DHEAD, DHEAD, 0, nullptr, nullptr, 0);
                gemm(Vb + off + h * DHEAD, HIDN, 0, Fp(44) + wi, DHEAD, nullptr,
                     VRb + off + h * DHEAD, HIDN, Mt[st], DHEAD, DHEAD, 0, nullptr, nullptr, 0);
            }
        }
        // attention: segment softmax over item-dst relations (shared) + user-dst
        hipMemsetAsync(MXI, 0, NIC * 2 * sizeof(unsigned), stream);
        hipMemsetAsync(SMI, 0, NIC * 2 * sizeof(float), stream);
        hipMemsetAsync(MXU, 0, NUC * 2 * sizeof(unsigned), stream);
        hipMemsetAsync(SMU, 0, NUC * 2 * sizeof(float), stream);
        size_t aoff = 0;
        for (int e = 0; e < 4; ++e) {
            hgt_alpha_kernel<<<g1((size_t)relE[e] * 2), 256, 0, stream>>>(
                Ip(relSi[e]), Ip(relDi[e]), Qb + toff[1] * HIDN, KRb + toff[relS[e]] * HIDN,
                Fp(45) + ((size_t)l * 5 + e) * NHEAD, EA + aoff * 2, MXI, relE[e], 0.125f);
            aoff += relE[e];
        }
        hgt_alpha_kernel<<<g1((size_t)EBC * 2), 256, 0, stream>>>(
            Ip(11), Ip(12), Qb + toff[0] * HIDN, KRb + toff[1] * HIDN,
            Fp(45) + ((size_t)l * 5 + 4) * NHEAD, EA + aoff * 2, MXU, EBC, 0.125f);
        size_t a2 = 0;
        for (int e = 0; e < 4; ++e) {
            seg_expsum_kernel<<<g1((size_t)relE[e] * 2), 256, 0, stream>>>(
                EA + a2 * 2, Ip(relDi[e]), MXI, EX + a2 * 2, SMI, relE[e], 2);
            a2 += relE[e];
        }
        seg_expsum_kernel<<<g1((size_t)EBC * 2), 256, 0, stream>>>(
            EA + a2 * 2, Ip(12), MXU, EX + a2 * 2, SMU, EBC, 2);
        // weighted message aggregation
        hipMemsetAsync(AGGI, 0, (size_t)NIC * HIDN * sizeof(float), stream);
        hipMemsetAsync(AGGU, 0, (size_t)NUC * HIDN * sizeof(float), stream);
        size_t a3 = 0;
        for (int e = 0; e < 4; ++e) {
            hgt_scatter_kernel<<<g1((size_t)relE[e] * HIDN), 256, 0, stream>>>(
                Ip(relSi[e]), Ip(relDi[e]), VRb + toff[relS[e]] * HIDN,
                EX + a3 * 2, SMI, AGGI, relE[e]);
            a3 += relE[e];
        }
        hgt_scatter_kernel<<<g1((size_t)EBC * HIDN), 256, 0, stream>>>(
            Ip(11), Ip(12), VRb + toff[1] * HIDN, EX + a3 * 2, SMU, AGGU, EBC);
        // o = gelu(agg) @ aW + ab; out = sigmoid(skip)*o + (1-s)*x   (fused epilogue)
        gemm(AGGU, HIDN, 1, Fp(40) + ((size_t)l * 5 + 0) * HIDN * HIDN, HIDN,
             Fp(41) + ((size_t)l * 5 + 0) * HIDN, OU, HIDN, NUC, HIDN, HIDN,
             2, Fp(42) + (l * 5 + 0), USER, HIDN);
        gemm(AGGI, HIDN, 1, Fp(40) + ((size_t)l * 5 + 1) * HIDN * HIDN, HIDN,
             Fp(41) + ((size_t)l * 5 + 1) * HIDN, TMPI, HIDN, NIC, HIDN, HIDN,
             2, Fp(42) + (l * 5 + 1), ITEM, HIDN);
        // residual BN + relu
        bnstats_kernel<<<dim3(HIDN), 256, 0, stream>>>(OU, USER, 0, NUC, MV);
        bnapply_kernel<<<g1((size_t)NUC * HIDN), 256, 0, stream>>>(
            OU, USER, 0, MV, Fp(46) + ((size_t)l * 2 + 0) * HIDN,
            Fp(47) + ((size_t)l * 2 + 0) * HIDN, USER, 1, (size_t)NUC * HIDN);
        bnstats_kernel<<<dim3(HIDN), 256, 0, stream>>>(TMPI, ITEM, 0, NIC, MV);
        bnapply_kernel<<<g1((size_t)NIC * HIDN), 256, 0, stream>>>(
            TMPI, ITEM, 0, MV, Fp(46) + ((size_t)l * 2 + 1) * HIDN,
            Fp(47) + ((size_t)l * 2 + 1) * HIDN, ITEM, 1, (size_t)NIC * HIDN);
    }

    // ---- 4. final BN + relu per output ------------------------------------
    float* finP[5] = { USER, ITEM, TASTE, INTENT, IMAGE };
    int    finN[5] = { NUC, NIC, NIC, NIC, NIC };
    for (int i = 0; i < 5; ++i) {
        bnstats_kernel<<<dim3(HIDN), 256, 0, stream>>>(finP[i], nullptr, 0, finN[i], MV);
        bnapply_kernel<<<g1((size_t)finN[i] * HIDN), 256, 0, stream>>>(
            finP[i], nullptr, 0, MV, Fp(48) + (size_t)i * HIDN, Fp(49) + (size_t)i * HIDN,
            finP[i], 1, (size_t)finN[i] * HIDN);
    }
}
